// model_9251359556171
// MI455X (gfx1250) — matrix operand order repaired and checked by reading, not compiled, not run on hardware
//
#include <hip/hip_runtime.h>

// GraphSAGE (3x SAGEConv mean-aggregator with edge weights) + 2 dense layers.
// N=50000 nodes, E=800000 edges, F=128 features throughout (IN_F==HID==128).
//
// Strategy:
//   * edge scatter-add with hardware f32 atomics (one wave / edge, float4 / lane)
//   * all [N,128]x[128,128] GEMMs on v_wmma_f32_16x16x32_f16:
//       block = 16 rows x 128 cols, 8 waves, 1 wave = one 16x16 tile,
//       K=128 -> 4 chunks, self+neigh chained into one f32 accumulator (8 WMMAs)
//   * HAS_NEIGH is a template param -> branch-free WMMA chain; all fragments
//     preloaded so one load clause + one wait feeds 8 back-to-back WMMAs
//   * A tiles staged f32->f16 into LDS in WMMA A-fragment order (contiguous per lane)
//   * weights pre-packed f32->f16 into WMMA B-fragment order (contiguous per lane)
// Workspace: deg[N] + 2x N*128 f32 ping-pong + 7x 16K f16 packed weights ~= 52 MB.

typedef __attribute__((ext_vector_type(16))) _Float16 v16h;
typedef __attribute__((ext_vector_type(8)))  float    v8f;

#define F 128

// ------------------------------------------------------------------ utilities
__global__ void zero_f32(float* __restrict__ p, long long n) {
  long long i = (long long)blockIdx.x * blockDim.x + threadIdx.x;
  if (i < n) p[i] = 0.0f;
}

__global__ void degree_kernel(const int* __restrict__ dst, float* __restrict__ deg, int E) {
  int e = blockIdx.x * blockDim.x + threadIdx.x;
  if (e < E) unsafeAtomicAdd(&deg[dst[e]], 1.0f);
}

// Pack a 128x128 row-major fp32 weight (k,n) into WMMA-B fragment order fp16:
// flat index [ct(8)][c(4)][lane(32)][hh(16)], so each lane's 16-half B fragment
// for (coltile ct, kchunk c) is one contiguous 32-byte load.
// B layout (16-bit, 32x16): lane half g, half hh -> K = 16*(hh/8) + 8*g + (hh%8)
// (matches A-fragment order); column n = lane%16.
__global__ void pack_weight(const float* __restrict__ w, _Float16* __restrict__ out) {
  int p = blockIdx.x * blockDim.x + threadIdx.x;   // 0..16383
  int hh   = p & 15;
  int lane = (p >> 4) & 31;
  int c    = (p >> 9) & 3;
  int ct   = p >> 11;
  int k = c * 32 + (hh >> 3) * 16 + (lane >> 4) * 8 + (hh & 7);
  int n = ct * 16 + (lane & 15);
  out[p] = (_Float16)w[k * F + n];
}

// ------------------------------------------------------------- edge aggregation
// s[dst] += ew * h[src]; one wave per edge, 4 floats per lane (coalesced 512B).
__global__ __launch_bounds__(256) void aggregate_kernel(
    const float* __restrict__ h, const float* __restrict__ ew,
    const int* __restrict__ src, const int* __restrict__ dst,
    float* __restrict__ s, int E) {
  int wave = (int)((blockIdx.x * 256 + threadIdx.x) >> 5);
  int lane = threadIdx.x & 31;
  if (wave >= E) return;
  int sv = src[wave], dv = dst[wave];
  float w = ew[wave];
  float4 v = ((const float4*)(h + (size_t)sv * F))[lane];
  float* sp = s + (size_t)dv * F + lane * 4;
  unsafeAtomicAdd(sp + 0, v.x * w);
  unsafeAtomicAdd(sp + 1, v.y * w);
  unsafeAtomicAdd(sp + 2, v.z * w);
  unsafeAtomicAdd(sp + 3, v.w * w);
}

// ------------------------------------------------------------------ WMMA GEMM
// hout = relu( hin @ Wself [+ (s/max(deg,1)) @ Wneigh] + bias )
// Block: 16 rows x 128 cols. Wave wv owns column tile [16*wv, 16*wv+16).
// A-fragment swizzle (16-bit A 16x32): lane L = g*16+m (g from K), half hh:
//   K = 16*(hh/8) + 8*g + (hh%8)  => inverse used during LDS staging.
template <bool HAS_NEIGH>
__global__ __launch_bounds__(256) void sage_gemm(
    const float* __restrict__ hin, const float* __restrict__ s,
    const float* __restrict__ deg,
    const _Float16* __restrict__ wself, const _Float16* __restrict__ wneigh,
    const float* __restrict__ bias, float* __restrict__ hout,
    int nrows, int relu) {
  __shared__ alignas(32) _Float16 ldsH[2048];   // 16x128 f16, A-fragment order
  __shared__ alignas(32) _Float16 ldsN[2048];
  const int tid = threadIdx.x;
  const int rowbase = blockIdx.x * 16;

  // ---- stage: each thread converts 8 consecutive k of one row into LDS
  {
    const int m  = tid >> 4;          // 0..15 row-in-tile
    const int kb = (tid & 15) << 3;   // 0..120
    int row = rowbase + m; if (row >= nrows) row = nrows - 1;
    const float* hp = hin + (size_t)row * F + kb;
    const float* sp = nullptr; float inv = 0.0f;
    if (HAS_NEIGH) { sp = s + (size_t)row * F + kb; inv = 1.0f / fmaxf(deg[row], 1.0f); }
#pragma unroll
    for (int i = 0; i < 8; ++i) {
      int k = kb + i;
      int c = k >> 5, r = k & 31;
      int hh = (r >> 4) * 8 + (r & 7);
      int L  = ((r >> 3) & 1) * 16 + m;
      int idx = (c * 32 + L) * 16 + hh;
      ldsH[idx] = (_Float16)hp[i];
      if (HAS_NEIGH) ldsN[idx] = (_Float16)(sp[i] * inv);
    }
  }
  __syncthreads();

  // ---- compute: wave = one 16x16 tile; preload all fragments, then 8 WMMAs
  const int lane = tid & 31;
  const int wv   = tid >> 5;          // column tile 0..7

  v16h aH[4], bS[4], aN[4], bN[4];
#pragma unroll
  for (int c = 0; c < 4; ++c) {
    aH[c] = *(const v16h*)(ldsH + (c * 32 + lane) * 16);
    bS[c] = *(const v16h*)(wself + ((size_t)(wv * 4 + c) * 32 + lane) * 16);
    if (HAS_NEIGH) {
      aN[c] = *(const v16h*)(ldsN + (c * 32 + lane) * 16);
      bN[c] = *(const v16h*)(wneigh + ((size_t)(wv * 4 + c) * 32 + lane) * 16);
    }
  }

  v8f acc = {};
#pragma unroll
  for (int c = 0; c < 4; ++c) {
    acc = __builtin_amdgcn_wmma_f32_16x16x32_f16(false, aH[c], false, bS[c],
                                                 (short)0, acc, false, false);
    if (HAS_NEIGH)
      acc = __builtin_amdgcn_wmma_f32_16x16x32_f16(false, aN[c], false, bN[c],
                                                   (short)0, acc, false, false);
  }

  // ---- epilogue: C/D layout -> lane {g=lane/16, n=lane&15}, VGPR r -> row r+8g
  const int g = lane >> 4, nn = lane & 15;
  const float bv = bias[wv * 16 + nn];
#pragma unroll
  for (int r = 0; r < 8; ++r) {
    int orow = rowbase + r + 8 * g;
    if (orow < nrows) {
      float v = acc[r] + bv;
      if (relu) v = fmaxf(v, 0.0f);
      hout[(size_t)orow * F + wv * 16 + nn] = v;
    }
  }
}

// ----------------------------------------------------------------- final GEMV
// out[i] = h[i,:] . w + b ; one wave per node, float4/lane + shuffle reduction.
__global__ __launch_bounds__(256) void final_gemv(
    const float* __restrict__ h, const float* __restrict__ w,
    const float* __restrict__ b, float* __restrict__ out, int n) {
  int wave = (int)((blockIdx.x * 256 + threadIdx.x) >> 5);
  int lane = threadIdx.x & 31;
  if (wave >= n) return;
  float4 hv = ((const float4*)(h + (size_t)wave * F))[lane];
  float4 wv = ((const float4*)w)[lane];
  float a = hv.x * wv.x + hv.y * wv.y + hv.z * wv.z + hv.w * wv.w;
#pragma unroll
  for (int off = 16; off > 0; off >>= 1) a += __shfl_down(a, off, 32);
  if (lane == 0) out[wave] = a + b[0];
}

// --------------------------------------------------------------------- driver
extern "C" void kernel_launch(void* const* d_in, const int* in_sizes, int n_in,
                              void* d_out, int out_size, void* d_ws, size_t ws_size,
                              hipStream_t stream) {
  const float* inputs = (const float*)d_in[0];
  const float* ew     = (const float*)d_in[1];
  const int*   src    = (const int*)d_in[2];
  const int*   dst    = (const int*)d_in[3];
  const float* wS[3]  = {(const float*)d_in[4], (const float*)d_in[7],  (const float*)d_in[10]};
  const float* wN[3]  = {(const float*)d_in[5], (const float*)d_in[8],  (const float*)d_in[11]};
  const float* bb[3]  = {(const float*)d_in[6], (const float*)d_in[9],  (const float*)d_in[12]};
  const float* wlin1  = (const float*)d_in[13];
  const float* blin1  = (const float*)d_in[14];
  const float* wlin2  = (const float*)d_in[15];
  const float* blin2  = (const float*)d_in[16];
  float* out = (float*)d_out;

  const int N = in_sizes[0] / F;
  const int E = in_sizes[1];
  (void)n_in; (void)out_size; (void)ws_size;

  // carve workspace (hipMalloc base is 256B-aligned)
  char* p = (char*)d_ws;
  auto carve = [&](size_t bytes) { char* r = p; p += (bytes + 255) & ~(size_t)255; return r; };
  float*    deg = (float*)carve((size_t)N * 4);
  float*    B1  = (float*)carve((size_t)N * F * 4);
  float*    B2  = (float*)carve((size_t)N * F * 4);
  _Float16* wp[7];
  for (int i = 0; i < 7; ++i) wp[i] = (_Float16*)carve((size_t)16384 * 2);

  // pack all weight matrices to f16 B-fragment order
  pack_weight<<<64, 256, 0, stream>>>(wS[0], wp[0]);
  pack_weight<<<64, 256, 0, stream>>>(wN[0], wp[1]);
  pack_weight<<<64, 256, 0, stream>>>(wS[1], wp[2]);
  pack_weight<<<64, 256, 0, stream>>>(wN[1], wp[3]);
  pack_weight<<<64, 256, 0, stream>>>(wS[2], wp[4]);
  pack_weight<<<64, 256, 0, stream>>>(wN[2], wp[5]);
  pack_weight<<<64, 256, 0, stream>>>(wlin1, wp[6]);

  // in-degree (shared by all three layers)
  zero_f32<<<(N + 255) / 256, 256, 0, stream>>>(deg, N);
  degree_kernel<<<(E + 255) / 256, 256, 0, stream>>>(dst, deg, E);

  const long long NF = (long long)N * F;
  const int zgrid = (int)((NF + 255) / 256);
  const int agrid = (E + 7) / 8;        // 8 edges (waves) per block
  const int ggrid = (N + 15) / 16;      // 16 rows per block

  // layer 1: inputs -> B1   (aggregate into B2)
  zero_f32<<<zgrid, 256, 0, stream>>>(B2, NF);
  aggregate_kernel<<<agrid, 256, 0, stream>>>(inputs, ew, src, dst, B2, E);
  sage_gemm<true><<<ggrid, 256, 0, stream>>>(inputs, B2, deg, wp[0], wp[1], bb[0], B1, N, 1);

  // layer 2: B1 -> B2  (gemm safely overwrites its own agg input: rows staged to LDS first)
  zero_f32<<<zgrid, 256, 0, stream>>>(B2, NF);
  aggregate_kernel<<<agrid, 256, 0, stream>>>(B1, ew, src, dst, B2, E);
  sage_gemm<true><<<ggrid, 256, 0, stream>>>(B1, B2, deg, wp[2], wp[3], bb[1], B2, N, 1);

  // layer 3: B2 -> B1
  zero_f32<<<zgrid, 256, 0, stream>>>(B1, NF);
  aggregate_kernel<<<agrid, 256, 0, stream>>>(B2, ew, src, dst, B1, E);
  sage_gemm<true><<<ggrid, 256, 0, stream>>>(B2, B1, deg, wp[4], wp[5], bb[2], B1, N, 1);

  // lin1 (no neighbor term): B1 -> B2
  sage_gemm<false><<<ggrid, 256, 0, stream>>>(B1, nullptr, deg, wp[6], nullptr, blin1, B2, N, 1);

  // lin2: B2 -> out
  final_gemv<<<(N + 7) / 8, 256, 0, stream>>>(B2, wlin2, blin2, out, N);
}
